// MDL_RNN_mnist_19980187861906
// MI455X (gfx1250) — matrix operand order repaired and checked by reading, not compiled, not run on hardware
//
#include <hip/hip_runtime.h>
#include <math.h>

typedef __attribute__((ext_vector_type(16))) _Float16 v16h;
typedef __attribute__((ext_vector_type(8)))  float    v8f;

#define HD      2048
#define PM      64
#define NIN     784
#define NINP    800      // padded to multiple of 32 for WMMA K loop
#define NOUTC   10
#define BATCH   256
#define TSTEPS  100
#define TCHUNK  10
#define NBLK    (HD / 256)   // 8 neuron-blocks per batch row

#define DTC     1e-3f
#define VTHR    1.0f
#define TREF    (5.0f * DTC)
#define DT_TAUD (1e-3f / 0.03f)

// ---------------- deterministic counter hash -> U[0,1) -------------------
__device__ __forceinline__ float u01_hash(unsigned t, unsigned b, unsigned n) {
    unsigned h = t * 0x9E3779B9u ^ (b + 0x85EBCA6Bu) * 0xC2B2AE35u ^ (n * 0x27D4EB2Fu);
    h ^= h >> 16; h *= 0x7FEB352Du;
    h ^= h >> 15; h *= 0x846CA68Bu;
    h ^= h >> 16;
    return (float)(h >> 8) * (1.0f / 16777216.0f);
}

// ---------------- state init ----------------------------------------------
__global__ void k_init(float* I, float* mem, float* s, float* r,
                       float* tlast, float* rm, float* part0, float* part1) {
    int i = blockIdx.x * blockDim.x + threadIdx.x;
    I[i] = 0.f; mem[i] = 0.f; s[i] = 0.f; r[i] = 0.f;
    tlast[i] = -1.f; rm[i] = 0.f;
    if (i < BATCH * NBLK * PM) { part0[i] = 0.f; part1[i] = 0.f; }
}

// ---------------- Win f32 -> f16, pad K 784 -> 800 ------------------------
__global__ void k_cvt_win(const float* __restrict__ Win, _Float16* __restrict__ Winh) {
    int i = blockIdx.x * blockDim.x + threadIdx.x;        // H*NINP total
    int h = i / NINP, k = i % NINP;
    Winh[i] = (k < NIN) ? (_Float16)Win[h * NIN + k] : (_Float16)0.0f;
}

// ---------------- fold l into pout: poutl[h,p] = pout[h,p] * l[p] ---------
__global__ void k_cvt_pout(const float* __restrict__ pout, const float* __restrict__ l,
                           float* __restrict__ poutl) {
    int i = blockIdx.x * blockDim.x + threadIdx.x;        // H*PM total
    poutl[i] = pout[i] * l[i & (PM - 1)];
}

// ---------------- spike generation for one time chunk ---------------------
__global__ void k_spikes(const float* __restrict__ x, _Float16* __restrict__ spk, int t0) {
    int i = blockIdx.x * blockDim.x + threadIdx.x;        // TCHUNK*BATCH*NINP total
    int k  = i % NINP;
    int bt = i / NINP;
    int b  = bt % BATCH;
    int tl = bt / BATCH;
    _Float16 v = (_Float16)0.0f;
    if (k < NIN) {
        float u = u01_hash((unsigned)(t0 + tl), (unsigned)b, (unsigned)k);
        v = (u < x[b * NIN + k]) ? (_Float16)1.0f : (_Float16)0.0f;
    }
    spk[i] = v;
}

// ---------------- WMMA GEMM: Fin[tb,h] = sum_k spk[tb,k] * Win[h,k] -------
// A = spk chunk (M = TCHUNK*BATCH, K = NINP, f16 row-major, zero padded)
// B = Win^T via Winh (H x NINP f16 row-major) -> B[k,n] = Winh[n,k]
// One wave computes one 16x16 f32 tile with 25x v_wmma_f32_16x16x32_f16.
__global__ void k_gemm(const _Float16* __restrict__ spk,
                       const _Float16* __restrict__ winh,
                       float* __restrict__ fin) {
    const int NT   = HD / 16;                             // 128 tiles in N
    int wave = (int)(blockIdx.x * (blockDim.x >> 5)) + ((int)threadIdx.x >> 5);
    int lane = (int)threadIdx.x & 31;
    int tm = wave / NT;
    int tn = wave % NT;
    int half = lane >> 4;
    int l16  = lane & 15;

    const _Float16* arow = spk  + (size_t)(tm * 16 + l16) * NINP;  // A row m
    const _Float16* brow = winh + (size_t)(tn * 16 + l16) * NINP;  // B col n

    v8f acc = {};
    for (int k0 = 0; k0 < NINP; k0 += 32) {
        if (k0 + 32 < NINP) {                              // prefetch next K slab
            __builtin_prefetch(arow + k0 + 32, 0, 0);
            __builtin_prefetch(brow + k0 + 32, 0, 0);
        }
        union { unsigned u[8]; v16h h; } A, B;
        #pragma unroll
        for (int v = 0; v < 8; ++v) {
            // ISA 7.12.2: 16-bit A 16x32 layout (K pairs per VGPR, halves swap K groups)
            int ka = ((v < 4) ? (2 * v) : (16 + 2 * (v - 4))) + (half ? 8 : 0);
            // B 32x16 layout: same K order as A (K pairs per VGPR, halves swap K groups)
            int kb = ((v < 4) ? (2 * v) : (16 + 2 * (v - 4))) + (half ? 8 : 0);
            A.u[v] = *(const unsigned*)(arow + k0 + ka);
            B.u[v] = *(const unsigned*)(brow + k0 + kb);
        }
        acc = __builtin_amdgcn_wmma_f32_16x16x32_f16(
                  false, A.h, false, B.h, (short)0, acc, false, false);
    }
    // D layout: lane -> n = lane&15 ; VGPR i -> m = (lane<16 ? i : 8+i)
    float* dbase = fin + (size_t)(tm * 16 + half * 8) * HD + tn * 16 + l16;
    #pragma unroll
    for (int i = 0; i < 8; ++i)
        dbase[(size_t)i * HD] = acc[i];
}

// ---------------- fused LIF step ------------------------------------------
// Per block: 256 neurons of one batch row.
//  1) a[b,p] = sum over NBLK per-block partials (computed last step)
//  2) rec + LIF pointwise update, r_new
//  3) this block's partial for the NEXT step: part_out[b][blk][p] =
//     sum_{h in block} r_new[h] * poutl[h,p]   (race-free slot, no atomics)
__global__ void k_step(const float* __restrict__ fin_t,
                       const float* __restrict__ part_in, float* __restrict__ part_out,
                       const float* __restrict__ pin, const float* __restrict__ poutl,
                       float* __restrict__ I, float* __restrict__ mem,
                       float* __restrict__ s, float* __restrict__ r,
                       float* __restrict__ tlast, float* __restrict__ rm,
                       float tval, float ls, float lm, float ld) {
    __shared__ float av[PM];
    __shared__ float rv[256];
    __shared__ float red[256];
    int tid = (int)threadIdx.x;
    int idx = (int)(blockIdx.x * blockDim.x) + tid;
    int b = idx >> 11;                    // block-uniform (256 | 2048)
    int h = idx & (HD - 1);
    int hstart = h & ~255;                // first neuron of this block
    int gblk = hstart >> 8;               // 0..7

    // 1) gather partials from previous step
    if (tid < PM) {
        const float* pi = part_in + ((size_t)b * NBLK) * PM + tid;
        float sum = 0.f;
        #pragma unroll
        for (int g = 0; g < NBLK; ++g) sum += pi[g * PM];
        av[tid] = sum;
    }
    __syncthreads();

    // 2) recurrent input + LIF update
    const float4* pr = (const float4*)(pin + (size_t)h * PM);
    float rec = 0.f;
    #pragma unroll
    for (int q = 0; q < PM / 4; ++q) {
        float4 v = pr[q];
        rec += v.x * av[4 * q] + v.y * av[4 * q + 1]
             + v.z * av[4 * q + 2] + v.w * av[4 * q + 3];
    }

    float Iv = ls * I[idx] + fin_t[idx] + rec;           // rec uses r from prev step
    float sv = s[idx];
    float tl = tlast[idx];
    float refr = (tval > tl + TREF) ? 1.f : 0.f;
    float mv = refr * (lm * mem[idx] + (1.f - lm) * Iv) * (1.f - sv);
    float rvnew = ld * r[idx] + DT_TAUD * sv;
    float snew = (mv - VTHR > 0.f) ? 1.f : 0.f;
    tl = tl + (tval - tl) * snew;

    I[idx] = Iv; mem[idx] = mv; s[idx] = snew; r[idx] = rvnew; tlast[idx] = tl;
    float m0 = rm[idx];
    rm[idx] = rvnew > m0 ? rvnew : m0;

    // 3) partial of r_new @ poutl for the next step
    rv[tid] = rvnew;
    __syncthreads();
    int p = tid & 63, q = tid >> 6;       // 4 quarter-groups over 64 local rows each
    const float* po = poutl + (size_t)(hstart + q * 64) * PM + p;
    float acc = 0.f;
    #pragma unroll 8
    for (int j = 0; j < 64; ++j)
        acc += rv[q * 64 + j] * po[(size_t)j * PM];
    red[tid] = acc;
    __syncthreads();
    if (q == 0)
        part_out[((size_t)b * NBLK + gblk) * PM + p] =
            red[p] + red[p + 64] + red[p + 128] + red[p + 192];
}

// ---------------- readout: softmax(rm @ Wout^T) ---------------------------
__global__ void k_out(const float* __restrict__ rm, const float* __restrict__ Wout,
                      float* __restrict__ out) {
    __shared__ float red[NOUTC][257];
    int b = blockIdx.x, tid = threadIdx.x;
    float acc[NOUTC];
    #pragma unroll
    for (int o = 0; o < NOUTC; ++o) acc[o] = 0.f;
    for (int h = tid; h < HD; h += 256) {
        float v = rm[b * HD + h];
        #pragma unroll
        for (int o = 0; o < NOUTC; ++o) acc[o] += v * Wout[o * HD + h];
    }
    #pragma unroll
    for (int o = 0; o < NOUTC; ++o) red[o][tid] = acc[o];
    __syncthreads();
    for (int st = 128; st > 0; st >>= 1) {
        if (tid < st) {
            #pragma unroll
            for (int o = 0; o < NOUTC; ++o) red[o][tid] += red[o][tid + st];
        }
        __syncthreads();
    }
    if (tid == 0) {
        float mx = red[0][0];
        #pragma unroll
        for (int o = 1; o < NOUTC; ++o) mx = fmaxf(mx, red[o][0]);
        float e[NOUTC], sum = 0.f;
        #pragma unroll
        for (int o = 0; o < NOUTC; ++o) { e[o] = __expf(red[o][0] - mx); sum += e[o]; }
        float inv = 1.f / sum;
        #pragma unroll
        for (int o = 0; o < NOUTC; ++o) out[b * NOUTC + o] = e[o] * inv;
    }
}

// --------------------------------------------------------------------------
extern "C" void kernel_launch(void* const* d_in, const int* in_sizes, int n_in,
                              void* d_out, int out_size, void* d_ws, size_t ws_size,
                              hipStream_t stream) {
    const float* x    = (const float*)d_in[0];
    const float* Win  = (const float*)d_in[1];
    const float* Wout = (const float*)d_in[2];
    const float* pin  = (const float*)d_in[3];
    const float* pout = (const float*)d_in[4];
    const float* l    = (const float*)d_in[5];

    char* ws = (char*)d_ws;
    size_t off = 0;
    auto carve = [&](size_t bytes) -> void* {
        void* p = ws + off;
        off = (off + bytes + 255) & ~(size_t)255;
        return p;
    };
    float*    I     = (float*)carve((size_t)BATCH * HD * 4);
    float*    mem   = (float*)carve((size_t)BATCH * HD * 4);
    float*    s     = (float*)carve((size_t)BATCH * HD * 4);
    float*    r     = (float*)carve((size_t)BATCH * HD * 4);
    float*    tlast = (float*)carve((size_t)BATCH * HD * 4);
    float*    rm    = (float*)carve((size_t)BATCH * HD * 4);
    float*    part0 = (float*)carve((size_t)BATCH * NBLK * PM * 4);
    float*    part1 = (float*)carve((size_t)BATCH * NBLK * PM * 4);
    float*    poutl = (float*)carve((size_t)HD * PM * 4);
    _Float16* Winh  = (_Float16*)carve((size_t)HD * NINP * 2);
    _Float16* spk   = (_Float16*)carve((size_t)TCHUNK * BATCH * NINP * 2);
    float*    fin   = (float*)carve((size_t)TCHUNK * BATCH * HD * 4);

    const float ls = expf(-DTC / 0.01f);
    const float lm = expf(-DTC / 0.02f);
    const float ld = expf(-DTC / 0.03f);

    k_init<<<(BATCH * HD) / 256, 256, 0, stream>>>(I, mem, s, r, tlast, rm, part0, part1);
    k_cvt_win<<<(HD * NINP) / 256, 256, 0, stream>>>(Win, Winh);
    k_cvt_pout<<<(HD * PM) / 256, 256, 0, stream>>>(pout, l, poutl);

    float* pa = part0;   // partials produced at step t-1, consumed at step t
    float* pb = part1;

    for (int c = 0; c < TSTEPS / TCHUNK; ++c) {
        int t0 = c * TCHUNK;
        k_spikes<<<(TCHUNK * BATCH * NINP) / 256, 256, 0, stream>>>(x, spk, t0);
        // (M/16)*(N/16) tiles, 8 waves (one tile each) per 256-thread block
        int tiles = ((TCHUNK * BATCH) / 16) * (HD / 16);
        k_gemm<<<tiles / 8, 256, 0, stream>>>(spk, Winh, fin);
        for (int tl_ = 0; tl_ < TCHUNK; ++tl_) {
            float tval = DTC * (float)(t0 + tl_);
            k_step<<<(BATCH * HD) / 256, 256, 0, stream>>>(
                fin + (size_t)tl_ * BATCH * HD, pa, pb, pin, poutl,
                I, mem, s, r, tlast, rm, tval, ls, lm, ld);
            float* tswap = pa; pa = pb; pb = tswap;
        }
    }
    k_out<<<BATCH, 256, 0, stream>>>(rm, Wout, (float*)d_out);
}
